// FlowGNN_conv_block_76630806495924
// MI455X (gfx1250) — matrix operand order repaired and checked by reading, not compiled, not run on hardware
//
#include <hip/hip_runtime.h>
#include <hip/hip_bf16.h>

// ---------------------------------------------------------------------------
// FlowGNN conv block for MI455X (gfx1250, wave32, WMMA).
//   edge MLP (384->128 relu ->128)  via v_wmma_f32_16x16x32_bf16, M-blocked x2
//   scatter-sum fused via global f32 atomics
//   node MLP (256->128 relu ->128)  via v_wmma_f32_16x16x32_bf16
//   smoothing layer elementwise
// ---------------------------------------------------------------------------

typedef __bf16 bf16_t;
typedef __attribute__((ext_vector_type(16))) __bf16 v16bf;
typedef __attribute__((ext_vector_type(8)))  __bf16 v8bf;
typedef __attribute__((ext_vector_type(8)))  float  v8f;

union V16 { v16bf v; v8bf h[2]; };

#define NNODES 50000
#define NEDGES 500000
#define HDIM   128

static __device__ __forceinline__ v8f wmma_bf16(const V16& a, const v16bf& b, const v8f& c) {
    return __builtin_amdgcn_wmma_f32_16x16x32_bf16(false, a.v, false, b, (short)0, c,
                                                   false, false);
}

// ---------------------------------------------------------------------------
// Convert W[K][128] (f32, row-major) -> Wt[128][K] (bf16) so each WMMA lane's
// B-fragment (one output column, 16 consecutive K) is a contiguous 32B load.
// ---------------------------------------------------------------------------
__global__ void k_wtrans(const float* __restrict__ W, bf16_t* __restrict__ Wt, int K) {
    int idx = blockIdx.x * blockDim.x + threadIdx.x;
    if (idx < K * HDIM) {
        int k = idx >> 7;        // row in W
        int n = idx & 127;       // col in W
        Wt[n * K + k] = (bf16_t)W[k * HDIM + n];
    }
}

// ---------------------------------------------------------------------------
// Edge MLP: per block = 32 edges, 8 waves; each wave computes TWO 16x16 C
// tiles (edges 0-15 and 16-31) sharing every B fragment -> halves weight
// traffic per WMMA and gives 2 independent WMMAs per load group.
// ---------------------------------------------------------------------------
__global__ __launch_bounds__(256, 2)
void k_edge_mlp(const float* __restrict__ node_attr,
                const float* __restrict__ edge_attr,
                const int*   __restrict__ eidx,      // [2][E]: row0=src,row1=dst
                const bf16_t* __restrict__ We1t, const float* __restrict__ be1,
                const bf16_t* __restrict__ We2t, const float* __restrict__ be2,
                float* __restrict__ agg) {
    __shared__ alignas(32) bf16_t sEin[32][392];   // 384 + pad (rows stay 16B aligned)
    __shared__ alignas(32) bf16_t sH[32][136];     // 128 + pad
    __shared__ int sSrc[32], sDst[32];

    const int tid = threadIdx.x;
    const int e0  = blockIdx.x * 32;

    if (tid < 32) {
        sSrc[tid] = eidx[e0 + tid];
        sDst[tid] = eidx[NEDGES + e0 + tid];
    }
    __syncthreads();

    // Cooperative gather+convert with float4: 8 threads per row, 12 chunks each.
    {
        const int r      = tid >> 3;       // 0..31
        const int lane8  = tid & 7;
        const float4* psrc = (const float4*)(node_attr + (size_t)sSrc[r] * HDIM);
        const float4* pdst = (const float4*)(node_attr + (size_t)sDst[r] * HDIM);
        const float4* pe   = (const float4*)(edge_attr + (size_t)(e0 + r) * HDIM);
#pragma unroll
        for (int j = 0; j < 12; ++j) {
            const int c4 = lane8 + j * 8;  // 0..95 (4-col chunks; never straddles segs)
            float4 v;
            if (c4 < 32)       v = psrc[c4];
            else if (c4 < 64)  v = pdst[c4 - 32];
            else               v = pe[c4 - 64];
            bf16_t* o = &sEin[r][c4 * 4];
            o[0] = (bf16_t)v.x; o[1] = (bf16_t)v.y;
            o[2] = (bf16_t)v.z; o[3] = (bf16_t)v.w;
        }
    }
    __syncthreads();

    const int lane  = tid & 31;
    const int wave  = tid >> 5;
    const int nbase = wave * 16;
    const int m     = lane & 15;       // A row / C column-within-tile
    const int hi    = lane >> 4;       // lane half
    const int n     = nbase + m;       // output column
    const int ah    = hi * 16;         // A K-offset (contiguous k = 16*hi + s)
    const int bh    = hi * 16;         // B K-offset

    // ---------------- layer 1: [32x384] @ [384x128] ----------------
    v8f c0a = {}, c0b = {};
    const bf16_t* arow0 = &sEin[m][0];
    const bf16_t* arow1 = &sEin[16 + m][0];
    const bf16_t* bcol  = We1t + n * 384;
    __builtin_prefetch(bcol, 0, 3);
#pragma unroll
    for (int kt = 0; kt < 12; ++kt) {
        const int base = kt * 32;
        V16 a0, a1;
        a0.h[0] = *(const v8bf*)(arow0 + base + ah);
        a0.h[1] = *(const v8bf*)(arow0 + base + ah + 8);
        a1.h[0] = *(const v8bf*)(arow1 + base + ah);
        a1.h[1] = *(const v8bf*)(arow1 + base + ah + 8);
        const v16bf b = *(const v16bf*)(bcol + base + bh);
        c0a = wmma_bf16(a0, b, c0a);
        c0b = wmma_bf16(a1, b, c0b);
    }
    const float bias1 = be1[n];
#pragma unroll
    for (int r = 0; r < 8; ++r) {
        // C layout: element r -> row M = r + 8*hi, column N = n
        sH[r + hi * 8][n]      = (bf16_t)fmaxf(c0a[r] + bias1, 0.0f);
        sH[16 + r + hi * 8][n] = (bf16_t)fmaxf(c0b[r] + bias1, 0.0f);
    }
    __syncthreads();

    // ---------------- layer 2: [32x128] @ [128x128] ----------------
    v8f c1a = {}, c1b = {};
    const bf16_t* hrow0 = &sH[m][0];
    const bf16_t* hrow1 = &sH[16 + m][0];
    const bf16_t* bcol2 = We2t + n * 128;
#pragma unroll
    for (int kt = 0; kt < 4; ++kt) {
        const int base = kt * 32;
        V16 a0, a1;
        a0.h[0] = *(const v8bf*)(hrow0 + base + ah);
        a0.h[1] = *(const v8bf*)(hrow0 + base + ah + 8);
        a1.h[0] = *(const v8bf*)(hrow1 + base + ah);
        a1.h[1] = *(const v8bf*)(hrow1 + base + ah + 8);
        const v16bf b = *(const v16bf*)(bcol2 + base + bh);
        c1a = wmma_bf16(a0, b, c1a);
        c1b = wmma_bf16(a1, b, c1b);
    }
    const float bias2 = be2[n];
#pragma unroll
    for (int r = 0; r < 8; ++r) {
        const int rowA = r + hi * 8;               // edge (tile A) within block
        const int rowB = 16 + rowA;                // edge (tile B) within block
        atomicAdd(&agg[sDst[rowA] * HDIM + n], c1a[r] + bias2);
        atomicAdd(&agg[sDst[rowB] * HDIM + n], c1b[r] + bias2);
    }
}

// ---------------------------------------------------------------------------
// Node MLP: per block = 16 nodes. a_in = [x | agg] (16 x 256 bf16).
// ---------------------------------------------------------------------------
__global__ __launch_bounds__(256, 2)
void k_node_mlp(const float* __restrict__ node_attr,
                const float* __restrict__ agg,
                const bf16_t* __restrict__ Wn1t, const float* __restrict__ bn1,
                const bf16_t* __restrict__ Wn2t, const float* __restrict__ bn2,
                float* __restrict__ updated_nodes) {
    __shared__ alignas(32) bf16_t sAin[16][264];   // 256 + pad
    __shared__ alignas(32) bf16_t sH[16][136];

    const int tid  = threadIdx.x;
    const int row0 = blockIdx.x * 16;

    // float4 staging: 16 threads per row, 4 chunks each.
    {
        const int r      = tid >> 4;       // 0..15
        const int lane16 = tid & 15;
        const float4* px = (const float4*)(node_attr + (size_t)(row0 + r) * HDIM);
        const float4* pa = (const float4*)(agg       + (size_t)(row0 + r) * HDIM);
#pragma unroll
        for (int j = 0; j < 4; ++j) {
            const int c4 = lane16 + j * 16;               // 0..63
            const float4 v = (c4 < 32) ? px[c4] : pa[c4 - 32];
            bf16_t* o = &sAin[r][c4 * 4];
            o[0] = (bf16_t)v.x; o[1] = (bf16_t)v.y;
            o[2] = (bf16_t)v.z; o[3] = (bf16_t)v.w;
        }
    }
    __syncthreads();

    const int lane  = tid & 31;
    const int wave  = tid >> 5;
    const int nbase = wave * 16;
    const int m     = lane & 15;
    const int hi    = lane >> 4;
    const int n     = nbase + m;
    const int ah    = hi * 16;
    const int bh    = hi * 16;

    // ---------------- layer 1: K = 256 ----------------
    v8f c0 = {};
    const bf16_t* arow = &sAin[m][0];
    const bf16_t* bcol = Wn1t + n * 256;
    __builtin_prefetch(bcol, 0, 3);
#pragma unroll
    for (int kt = 0; kt < 8; ++kt) {
        const int base = kt * 32;
        V16 ua;
        ua.h[0] = *(const v8bf*)(arow + base + ah);
        ua.h[1] = *(const v8bf*)(arow + base + ah + 8);
        const v16bf b = *(const v16bf*)(bcol + base + bh);
        c0 = wmma_bf16(ua, b, c0);
    }
    const float bias1 = bn1[n];
#pragma unroll
    for (int r = 0; r < 8; ++r) {
        sH[r + hi * 8][n] = (bf16_t)fmaxf(c0[r] + bias1, 0.0f);
    }
    __syncthreads();

    // ---------------- layer 2: K = 128 ----------------
    v8f c1 = {};
    const bf16_t* arow2 = &sH[m][0];
    const bf16_t* bcol2 = Wn2t + n * 128;
#pragma unroll
    for (int kt = 0; kt < 4; ++kt) {
        const int base = kt * 32;
        V16 ua;
        ua.h[0] = *(const v8bf*)(arow2 + base + ah);
        ua.h[1] = *(const v8bf*)(arow2 + base + ah + 8);
        const v16bf b = *(const v16bf*)(bcol2 + base + bh);
        c1 = wmma_bf16(ua, b, c1);
    }
    const float bias2 = bn2[n];
#pragma unroll
    for (int r = 0; r < 8; ++r) {
        const int row = row0 + r + hi * 8;
        updated_nodes[row * HDIM + n] = c1[r] + bias2;
    }
}

// ---------------------------------------------------------------------------
// Smoothing: edges = 0.5*(un[src]+un[dst]); fused scatter-sum + degree count.
// ---------------------------------------------------------------------------
__global__ void k_smooth_edges(const float* __restrict__ un,
                               const int* __restrict__ eidx,
                               float* __restrict__ out_edges,
                               float* __restrict__ accum,
                               float* __restrict__ deg) {
    const int e = blockIdx.x;
    const int c = threadIdx.x;
    const int s = eidx[e];
    const int d = eidx[NEDGES + e];
    const float v = 0.5f * (un[s * HDIM + c] + un[d * HDIM + c]);
    out_edges[(size_t)e * HDIM + c] = v;
    atomicAdd(&accum[d * HDIM + c], v);
    if (c == 0) atomicAdd(&deg[d], 1.0f);
}

__global__ void k_smooth_nodes(const float* __restrict__ accum,
                               const float* __restrict__ deg,
                               float* __restrict__ out_nodes) {
    const int idx = blockIdx.x * blockDim.x + threadIdx.x;
    if (idx < NNODES * HDIM) {
        out_nodes[idx] = accum[idx] / fmaxf(deg[idx >> 7], 1.0f);
    }
}

// ---------------------------------------------------------------------------
// Host-side orchestration (graph-capture safe: only async ops on `stream`).
// ---------------------------------------------------------------------------
extern "C" void kernel_launch(void* const* d_in, const int* in_sizes, int n_in,
                              void* d_out, int out_size, void* d_ws, size_t ws_size,
                              hipStream_t stream) {
    const float* node_attr = (const float*)d_in[0];
    const float* edge_attr = (const float*)d_in[1];
    const int*   eidx      = (const int*)d_in[2];
    const float* We1 = (const float*)d_in[3];
    const float* be1 = (const float*)d_in[4];
    const float* We2 = (const float*)d_in[5];
    const float* be2 = (const float*)d_in[6];
    const float* Wn1 = (const float*)d_in[7];
    const float* bn1 = (const float*)d_in[8];
    const float* Wn2 = (const float*)d_in[9];
    const float* bn2 = (const float*)d_in[10];

    char* ws = (char*)d_ws;
    // Workspace layout (bytes)
    bf16_t* We1t = (bf16_t*)(ws + 0);          // 384*128*2 =  98304
    bf16_t* We2t = (bf16_t*)(ws + 98304);      // 128*128*2 =  32768
    bf16_t* Wn1t = (bf16_t*)(ws + 131072);     // 256*128*2 =  65536
    bf16_t* Wn2t = (bf16_t*)(ws + 196608);     // 128*128*2 =  32768
    float*  agg  = (float*)(ws + 262144);              // N*H*4 = 25.6 MB (reused as smooth accum)
    float*  un   = (float*)(ws + 262144 + 25600000);   // N*H*4 = 25.6 MB
    float*  deg  = (float*)(ws + 262144 + 51200000);   // N*4   = 0.2 MB

    float* out_nodes = (float*)d_out;
    float* out_edges = out_nodes + (size_t)NNODES * HDIM;

    // 1) weights -> bf16, transposed [N][K]
    k_wtrans<<<(384 * 128 + 255) / 256, 256, 0, stream>>>(We1, We1t, 384);
    k_wtrans<<<(128 * 128 + 255) / 256, 256, 0, stream>>>(We2, We2t, 128);
    k_wtrans<<<(256 * 128 + 255) / 256, 256, 0, stream>>>(Wn1, Wn1t, 256);
    k_wtrans<<<(128 * 128 + 255) / 256, 256, 0, stream>>>(Wn2, Wn2t, 128);

    // 2) edge MLP + fused segment_sum into agg
    hipMemsetAsync(agg, 0, (size_t)NNODES * HDIM * sizeof(float), stream);
    k_edge_mlp<<<NEDGES / 32, 256, 0, stream>>>(node_attr, edge_attr, eidx,
                                                We1t, be1, We2t, be2, agg);

    // 3) node MLP -> updated_nodes
    k_node_mlp<<<NNODES / 16, 256, 0, stream>>>(node_attr, agg,
                                                Wn1t, bn1, Wn2t, bn2, un);

    // 4) smoothing (reuse agg buffer as accumulator)
    hipMemsetAsync(agg, 0, (size_t)NNODES * HDIM * sizeof(float), stream);
    hipMemsetAsync(deg, 0, (size_t)NNODES * sizeof(float), stream);
    k_smooth_edges<<<NEDGES, HDIM, 0, stream>>>(un, eidx, out_edges, agg, deg);
    k_smooth_nodes<<<(NNODES * HDIM + 255) / 256, 256, 0, stream>>>(agg, deg, out_nodes);
}
